// moe_already_emb_16741782520582
// MI455X (gfx1250) — matrix operand order repaired and checked by reading, not compiled, not run on hardware
//
#include <hip/hip_runtime.h>
#include <hip/hip_bf16.h>

// ---------------- Model constants (Mixtral-style, scaled) ----------------
#define LNUM   2
#define HDIM   1024
#define NHEADS 16
#define KVHEADS 4
#define HEADD  64
#define FDIM   1024
#define ENUM   8
#define SEQ    2048
#define TOK    SEQ          // B = 1
#define EPSV   1e-5f
#define THETA  1e6f

// ---------------- Types for WMMA (wave32, CDNA5) ----------------
typedef __bf16  bf16x16 __attribute__((ext_vector_type(16)));
typedef float   f32x8   __attribute__((ext_vector_type(8)));
typedef unsigned int u32x4 __attribute__((ext_vector_type(4)));

union FragAB { bf16x16 v; u32x4 q[2]; };

__device__ __forceinline__ unsigned short f2bf(float f) {
    unsigned u = __builtin_bit_cast(unsigned, f);
    unsigned rnd = 0x7FFFu + ((u >> 16) & 1u);
    return (unsigned short)((u + rnd) >> 16);
}

__device__ __forceinline__ f32x8 wmma_bf16(const FragAB& a, const FragAB& b, f32x8 c) {
    return __builtin_amdgcn_wmma_f32_16x16x32_bf16(false, a.v, false, b.v,
                                                   (short)0, c, false, false);
}

// ---------------- Elementwise helpers ----------------
__global__ void k_f32_to_bf16(const float* __restrict__ in,
                              unsigned short* __restrict__ out, int n) {
    int i = blockIdx.x * blockDim.x + threadIdx.x;
    if (i < n) out[i] = f2bf(in[i]);
}

// RMSNorm over rows of [TOK, HDIM]; optional bf16 mirror.
__global__ void k_rmsnorm(const float* __restrict__ x, const float* __restrict__ w,
                          float* __restrict__ y, unsigned short* __restrict__ yb) {
    int row = blockIdx.x;
    const float* xr = x + (size_t)row * HDIM;
    __shared__ float red[256];
    float s = 0.f;
    for (int i = threadIdx.x; i < HDIM; i += 256) { float v = xr[i]; s += v * v; }
    red[threadIdx.x] = s; __syncthreads();
    for (int off = 128; off > 0; off >>= 1) {
        if (threadIdx.x < off) red[threadIdx.x] += red[threadIdx.x + off];
        __syncthreads();
    }
    float rinv = rsqrtf(red[0] / (float)HDIM + EPSV);
    for (int i = threadIdx.x; i < HDIM; i += 256) {
        float v = xr[i] * rinv * w[i];
        y[(size_t)row * HDIM + i] = v;
        if (yb) yb[(size_t)row * HDIM + i] = f2bf(v);
    }
}

// ------- WMMA GEMM: C[M,N] (+)= A[M,K] * Bt[N,K]^T, 32x64 tile per wave -------
// A, Bt are bf16 (raw u16). One wave computes a 32(M)x64(N) tile:
// 2 A-fragments x 4 B-fragments -> 8 chained WMMAs per 32-wide K step.
__global__ __launch_bounds__(32)
void k_gemm_bf16(const unsigned short* __restrict__ A,
                 const unsigned short* __restrict__ Bt,
                 float* __restrict__ C,
                 int M, int N, int Kd, int addFlag) {
    const int lane = threadIdx.x;
    const int half = lane >> 4, r = lane & 15;
    const int tm = blockIdx.x * 32;
    const int tn = blockIdx.y * 64;

    const unsigned short* arow[2];
    arow[0] = A + (size_t)(tm + r) * Kd;
    arow[1] = A + (size_t)(tm + 16 + r) * Kd;
    const unsigned short* brow[4];
    #pragma unroll
    for (int nj = 0; nj < 4; ++nj)
        brow[nj] = Bt + (size_t)(tn + nj * 16 + r) * Kd;

    f32x8 acc[2][4];
    #pragma unroll
    for (int mi = 0; mi < 2; ++mi)
        #pragma unroll
        for (int nj = 0; nj < 4; ++nj) acc[mi][nj] = (f32x8){};

    for (int k0 = 0; k0 < Kd; k0 += 32) {
        FragAB a[2], b[4];
        #pragma unroll
        for (int mi = 0; mi < 2; ++mi) {
            const u32x4* pa = (const u32x4*)(arow[mi] + k0 + half * 8);
            a[mi].q[0] = pa[0]; a[mi].q[1] = pa[2];   // K 0..7/16..23 (or +8)
            __builtin_prefetch(arow[mi] + k0 + 128, 0, 3);
        }
        #pragma unroll
        for (int nj = 0; nj < 4; ++nj) {
            const u32x4* pb = (const u32x4*)(brow[nj] + k0 + half * 8);
            b[nj].q[0] = pb[0]; b[nj].q[1] = pb[2];   // K 0..7/16..23 (or +8)
            __builtin_prefetch(brow[nj] + k0 + 128, 0, 3);
        }
        #pragma unroll
        for (int mi = 0; mi < 2; ++mi)
            #pragma unroll
            for (int nj = 0; nj < 4; ++nj)
                acc[mi][nj] = wmma_bf16(a[mi], b[nj], acc[mi][nj]);
    }

    #pragma unroll
    for (int mi = 0; mi < 2; ++mi) {
        float* crow = C + (size_t)(tm + mi * 16 + half * 8) * N + r;
        #pragma unroll
        for (int nj = 0; nj < 4; ++nj) {
            #pragma unroll
            for (int i = 0; i < 8; ++i) {
                size_t idx = (size_t)i * N + tn + nj * 16;
                float v = acc[mi][nj][i];
                if (addFlag) v += crow[idx];
                crow[idx] = v;
            }
        }
    }
}

// ---------------- RoPE + layout packing ----------------
// X: [S, nh*64] fp32 -> Out: [nh, S, 64] bf16 with RoPE applied.
__global__ void k_rope(const float* __restrict__ X, unsigned short* __restrict__ Out,
                       int nh) {
    int idx = blockIdx.x * blockDim.x + threadIdx.x;
    int total = nh * SEQ * HEADD;
    if (idx >= total) return;
    int d = idx & 63;
    int t = (idx >> 6) & (SEQ - 1);
    int h = idx >> 17;                     // / (SEQ*64)
    int i = (d < 32) ? d : d - 32;
    float invf = __powf(THETA, -(2.0f * (float)i) / 64.0f);
    float ang = (float)t * invf;
    float c = __cosf(ang), s = __sinf(ang);
    size_t base = (size_t)t * (nh * HEADD) + h * HEADD;
    float xv = X[base + d];
    float other = (d < 32) ? -X[base + d + 32] : X[base + d - 32];
    Out[((size_t)h * SEQ + t) * HEADD + d] = f2bf(xv * c + other * s);
}

// V: [S, KVHEADS*64] fp32 -> Vt: [KVHEADS, 64, S] bf16 (transposed).
__global__ void k_pack_vt(const float* __restrict__ V, unsigned short* __restrict__ Vt) {
    int idx = blockIdx.x * blockDim.x + threadIdx.x;
    int total = KVHEADS * SEQ * HEADD;
    if (idx >= total) return;
    int d = idx & 63;
    int t = (idx >> 6) & (SEQ - 1);
    int h = idx >> 17;
    float v = V[(size_t)t * (KVHEADS * HEADD) + h * HEADD + d];
    Vt[((size_t)h * HEADD + d) * SEQ + t] = f2bf(v);
}

// ---------------- Flash-style causal attention (WMMA) ----------------
// Qb:[NH,S,64] Kb:[KVH,S,64] Vt:[KVH,64,S] (bf16) -> Ob:[S,H] bf16.
__global__ __launch_bounds__(32)
void k_attn(const unsigned short* __restrict__ Qb,
            const unsigned short* __restrict__ Kb,
            const unsigned short* __restrict__ Vt,
            unsigned short* __restrict__ Ob) {
    const int lane = threadIdx.x, half = lane >> 4, r = lane & 15;
    const int qb = blockIdx.x * 16;
    const int h  = blockIdx.y;
    const int kvh = h / (NHEADS / KVHEADS);

    // Q A-fragments: hd 0..31 and 32..63.
    const unsigned short* qrow = Qb + ((size_t)h * SEQ + qb + r) * HEADD;
    FragAB aq0, aq1;
    { const u32x4* p  = (const u32x4*)(qrow + half * 8);
      aq0.q[0] = p[0];  aq0.q[1] = p[2];
      const u32x4* p2 = (const u32x4*)(qrow + 32 + half * 8);
      aq1.q[0] = p2[0]; aq1.q[1] = p2[2]; }

    float m[8], lsum[8];
    f32x8 o[4];
    #pragma unroll
    for (int i = 0; i < 8; ++i) { m[i] = -3e38f; lsum[i] = 0.f; }
    #pragma unroll
    for (int t = 0; t < 4; ++t) o[t] = (f32x8){};

    __shared__ unsigned short Pl[16 * 32];

    for (int kb = 0; kb <= qb + 15; kb += 32) {
        // K B-fragments: two key column tiles x two hd halves.
        FragAB b00, b01, b10, b11;
        { const unsigned short* kp = Kb + ((size_t)kvh * SEQ + kb + r) * HEADD + half * 8;
          const u32x4* p = (const u32x4*)kp;
          b00.q[0] = p[0]; b00.q[1] = p[2];
          const u32x4* p1 = (const u32x4*)(kp + 32);
          b01.q[0] = p1[0]; b01.q[1] = p1[2];
          const u32x4* p2 = (const u32x4*)(kp + 16 * HEADD);
          b10.q[0] = p2[0]; b10.q[1] = p2[2];
          const u32x4* p3 = (const u32x4*)(kp + 16 * HEADD + 32);
          b11.q[0] = p3[0]; b11.q[1] = p3[2]; }

        f32x8 s0 = {}, s1 = {};
        s0 = wmma_bf16(aq0, b00, s0); s0 = wmma_bf16(aq1, b01, s0);
        s1 = wmma_bf16(aq0, b10, s1); s1 = wmma_bf16(aq1, b11, s1);

        const int kn0 = kb + r, kn1 = kb + 16 + r;
        float sv0[8], sv1[8];
        #pragma unroll
        for (int i = 0; i < 8; ++i) {
            int row = qb + half * 8 + i;
            float v0 = s0[i] * 0.125f; if (kn0 > row) v0 = -3e38f;
            float v1 = s1[i] * 0.125f; if (kn1 > row) v1 = -3e38f;
            sv0[i] = v0; sv1[i] = v1;
        }
        // Online softmax: row reductions across the 16 lanes of each half.
        #pragma unroll
        for (int i = 0; i < 8; ++i) {
            float x = fmaxf(sv0[i], sv1[i]);
            for (int off = 8; off >= 1; off >>= 1)
                x = fmaxf(x, __shfl_xor(x, off, 16));
            float nm = fmaxf(m[i], x);
            float alpha = __expf(m[i] - nm);
            float p0 = __expf(sv0[i] - nm);
            float p1 = __expf(sv1[i] - nm);
            float rs = p0 + p1;
            for (int off = 8; off >= 1; off >>= 1)
                rs += __shfl_xor(rs, off, 16);
            lsum[i] = lsum[i] * alpha + rs;
            m[i] = nm;
            #pragma unroll
            for (int t = 0; t < 4; ++t) o[t][i] *= alpha;
            Pl[(half * 8 + i) * 32 + r]      = f2bf(p0);
            Pl[(half * 8 + i) * 32 + 16 + r] = f2bf(p1);
        }
        __syncthreads();
        // Reload P in A-fragment layout (16x32, contraction over keys).
        FragAB pa;
        { const u32x4* p = (const u32x4*)(Pl + r * 32 + half * 8);
          pa.q[0] = p[0]; pa.q[1] = p[2]; }
        // P @ V for four 16-wide hd tiles.
        #pragma unroll
        for (int t = 0; t < 4; ++t) {
            FragAB bv;
            const u32x4* p = (const u32x4*)(Vt + ((size_t)kvh * HEADD + t * 16 + r) * SEQ
                                            + kb + half * 8);
            bv.q[0] = p[0]; bv.q[1] = p[2];
            o[t] = wmma_bf16(pa, bv, o[t]);
        }
        __syncthreads();
    }

    #pragma unroll
    for (int t = 0; t < 4; ++t)
        #pragma unroll
        for (int i = 0; i < 8; ++i) {
            int row = qb + half * 8 + i;
            float val = o[t][i] / lsum[i];
            Ob[(size_t)row * HDIM + h * HEADD + t * 16 + r] = f2bf(val);
        }
}

// ---------------- MoE routing ----------------
__global__ void k_gate(const float* __restrict__ xn, const float* __restrict__ gw,
                       float* __restrict__ logits) {
    int idx = blockIdx.x * blockDim.x + threadIdx.x;
    if (idx >= TOK * ENUM) return;
    int e = idx % ENUM, t = idx / ENUM;
    const float* x = xn + (size_t)t * HDIM;
    const float* w = gw + (size_t)e * HDIM;
    float s = 0.f;
    for (int i = 0; i < HDIM; ++i) s += x[i] * w[i];
    logits[idx] = s;
}

__global__ void k_route(const float* __restrict__ logits, float* __restrict__ wfull) {
    int t = blockIdx.x * blockDim.x + threadIdx.x;
    if (t >= TOK) return;
    float l[ENUM]; float mx = -3e38f;
    for (int e = 0; e < ENUM; ++e) { l[e] = logits[t * ENUM + e]; mx = fmaxf(mx, l[e]); }
    float se = 0.f;
    for (int e = 0; e < ENUM; ++e) { l[e] = __expf(l[e] - mx); se += l[e]; }
    for (int e = 0; e < ENUM; ++e) l[e] /= se;
    int i1 = 0; float p1 = -1.f;
    for (int e = 0; e < ENUM; ++e) if (l[e] > p1) { p1 = l[e]; i1 = e; }
    int i2 = -1; float p2 = -1.f;
    for (int e = 0; e < ENUM; ++e) if (e != i1 && l[e] > p2) { p2 = l[e]; i2 = e; }
    float den = p1 + p2;
    for (int e = 0; e < ENUM; ++e)
        wfull[t * ENUM + e] = (e == i1) ? p1 / den : ((e == i2) ? p2 / den : 0.f);
}

// a = silu(g) * u * wfull[:, e]  -> bf16
__global__ void k_moe_act(const float* __restrict__ g, const float* __restrict__ u,
                          const float* __restrict__ wfull, int e,
                          unsigned short* __restrict__ ab) {
    int idx = blockIdx.x * blockDim.x + threadIdx.x;
    if (idx >= TOK * FDIM) return;
    int t = idx / FDIM;
    float gv = g[idx];
    float sil = gv / (1.f + __expf(-gv));
    ab[idx] = f2bf(sil * u[idx] * wfull[t * ENUM + e]);
}

// ---------------- Orchestration ----------------
extern "C" void kernel_launch(void* const* d_in, const int* in_sizes, int n_in,
                              void* d_out, int out_size, void* d_ws, size_t ws_size,
                              hipStream_t stream) {
    (void)in_sizes; (void)n_in; (void)out_size; (void)ws_size;
    const float* emb = (const float*)d_in[0];
    const float* ln1 = (const float*)d_in[1];
    const float* ln2 = (const float*)d_in[2];
    const float* fln = (const float*)d_in[3];
    const float* qw  = (const float*)d_in[4];
    const float* kw  = (const float*)d_in[5];
    const float* vw  = (const float*)d_in[6];
    const float* ow  = (const float*)d_in[7];
    const float* gw  = (const float*)d_in[8];
    const float* w1  = (const float*)d_in[9];
    const float* w2  = (const float*)d_in[10];
    const float* w3  = (const float*)d_in[11];
    float* out = (float*)d_out;

    char* ws = (char*)d_ws;
    size_t off = 0;
    auto alloc = [&](size_t bytes) -> void* {
        void* p = ws + off; off += (bytes + 255) & ~(size_t)255; return p;
    };
    const size_t QWN = (size_t)LNUM * HDIM * HDIM;            // 2M
    const size_t KWN = (size_t)LNUM * (KVHEADS*HEADD) * HDIM; // 512K
    const size_t MWN = (size_t)LNUM * ENUM * FDIM * HDIM;     // 16M

    unsigned short* wq_b = (unsigned short*)alloc(QWN * 2);
    unsigned short* wk_b = (unsigned short*)alloc(KWN * 2);
    unsigned short* wv_b = (unsigned short*)alloc(KWN * 2);
    unsigned short* wo_b = (unsigned short*)alloc(QWN * 2);
    unsigned short* w1_b = (unsigned short*)alloc(MWN * 2);
    unsigned short* w2_b = (unsigned short*)alloc(MWN * 2);
    unsigned short* w3_b = (unsigned short*)alloc(MWN * 2);

    float* hbuf = (float*)alloc((size_t)TOK * HDIM * 4);
    float* xn   = (float*)alloc((size_t)TOK * HDIM * 4);
    unsigned short* xn_b = (unsigned short*)alloc((size_t)TOK * HDIM * 2);
    float* qf   = (float*)alloc((size_t)TOK * HDIM * 4);
    float* kf   = (float*)alloc((size_t)TOK * KVHEADS * HEADD * 4);
    float* vf   = (float*)alloc((size_t)TOK * KVHEADS * HEADD * 4);
    unsigned short* Qb = (unsigned short*)alloc((size_t)NHEADS * SEQ * HEADD * 2);
    unsigned short* Kb = (unsigned short*)alloc((size_t)KVHEADS * SEQ * HEADD * 2);
    unsigned short* Vt = (unsigned short*)alloc((size_t)KVHEADS * SEQ * HEADD * 2);
    unsigned short* aob = (unsigned short*)alloc((size_t)TOK * HDIM * 2);
    float* gbuf = (float*)alloc((size_t)TOK * FDIM * 4);
    float* ubuf = (float*)alloc((size_t)TOK * FDIM * 4);
    unsigned short* ab = (unsigned short*)alloc((size_t)TOK * FDIM * 2);
    float* logits = (float*)alloc((size_t)TOK * ENUM * 4);
    float* wfull  = (float*)alloc((size_t)TOK * ENUM * 4);

    auto cvt = [&](const float* src, unsigned short* dst, size_t n) {
        k_f32_to_bf16<<<(unsigned)((n + 255) / 256), 256, 0, stream>>>(src, dst, (int)n);
    };
    cvt(qw, wq_b, QWN); cvt(kw, wk_b, KWN); cvt(vw, wv_b, KWN); cvt(ow, wo_b, QWN);
    cvt(w1, w1_b, MWN); cvt(w2, w2_b, MWN); cvt(w3, w3_b, MWN);

    hipMemcpyAsync(hbuf, emb, (size_t)TOK * HDIM * 4, hipMemcpyDeviceToDevice, stream);

    for (int l = 0; l < LNUM; ++l) {
        // ---- attention block ----
        k_rmsnorm<<<TOK, 256, 0, stream>>>(hbuf, ln1 + (size_t)l * HDIM, xn, xn_b);
        k_gemm_bf16<<<dim3(TOK/32, HDIM/64), 32, 0, stream>>>(
            xn_b, wq_b + (size_t)l * HDIM * HDIM, qf, TOK, HDIM, HDIM, 0);
        k_gemm_bf16<<<dim3(TOK/32, (KVHEADS*HEADD)/64), 32, 0, stream>>>(
            xn_b, wk_b + (size_t)l * KVHEADS * HEADD * HDIM, kf, TOK, KVHEADS*HEADD, HDIM, 0);
        k_gemm_bf16<<<dim3(TOK/32, (KVHEADS*HEADD)/64), 32, 0, stream>>>(
            xn_b, wv_b + (size_t)l * KVHEADS * HEADD * HDIM, vf, TOK, KVHEADS*HEADD, HDIM, 0);
        {
            int nq = NHEADS * SEQ * HEADD, nk = KVHEADS * SEQ * HEADD;
            k_rope<<<(nq + 255) / 256, 256, 0, stream>>>(qf, Qb, NHEADS);
            k_rope<<<(nk + 255) / 256, 256, 0, stream>>>(kf, Kb, KVHEADS);
            k_pack_vt<<<(nk + 255) / 256, 256, 0, stream>>>(vf, Vt);
        }
        k_attn<<<dim3(SEQ/16, NHEADS), 32, 0, stream>>>(Qb, Kb, Vt, aob);
        k_gemm_bf16<<<dim3(TOK/32, HDIM/64), 32, 0, stream>>>(
            aob, wo_b + (size_t)l * HDIM * HDIM, hbuf, TOK, HDIM, HDIM, 1);

        // ---- MoE block ----
        k_rmsnorm<<<TOK, 256, 0, stream>>>(hbuf, ln2 + (size_t)l * HDIM, xn, xn_b);
        k_gate<<<(TOK*ENUM + 255)/256, 256, 0, stream>>>(
            xn, gw + (size_t)l * ENUM * HDIM, logits);
        k_route<<<(TOK + 255)/256, 256, 0, stream>>>(logits, wfull);
        for (int e = 0; e < ENUM; ++e) {
            const size_t wofs = ((size_t)l * ENUM + e) * FDIM * HDIM;
            k_gemm_bf16<<<dim3(TOK/32, FDIM/64), 32, 0, stream>>>(
                xn_b, w1_b + wofs, gbuf, TOK, FDIM, HDIM, 0);
            k_gemm_bf16<<<dim3(TOK/32, FDIM/64), 32, 0, stream>>>(
                xn_b, w3_b + wofs, ubuf, TOK, FDIM, HDIM, 0);
            k_moe_act<<<(TOK*FDIM + 255)/256, 256, 0, stream>>>(gbuf, ubuf, wfull, e, ab);
            k_gemm_bf16<<<dim3(TOK/32, HDIM/64), 32, 0, stream>>>(
                ab, w2_b + wofs, hbuf, TOK, HDIM, FDIM, 1);
        }
    }
    k_rmsnorm<<<TOK, 256, 0, stream>>>(hbuf, fln, out, (unsigned short*)nullptr);
}
